// MoELayer_82446192214075
// MI455X (gfx1250) — matrix operand order repaired and checked by reading, not compiled, not run on hardware
//
#include <hip/hip_runtime.h>
#include <hip/hip_bf16.h>

// Problem dims (fixed by the reference).
#define TT 2048   // tokens = B*S
#define HH 512    // hidden
#define FF 2048   // ffn
#define EE 8      // experts

typedef __attribute__((ext_vector_type(16))) __bf16 v16bf;
typedef __attribute__((ext_vector_type(8)))  float  v8f;
typedef unsigned short u16;

union BF16Frag { uint4 q[2]; v16bf v; };

__device__ __forceinline__ u16 f2bf(float f) {
  unsigned int u = __float_as_uint(f);
  u += 0x7FFFu + ((u >> 16) & 1u);          // round-to-nearest-even
  return (u16)(u >> 16);
}

__device__ __forceinline__ v8f wmma_bf16(v16bf a, v16bf b, v8f c) {
  // (neg_a, A, neg_b, B, c_mod, C, reuse_a, reuse_b)
  return __builtin_amdgcn_wmma_f32_16x16x32_bf16(false, a, false, b, (short)0, c,
                                                 false, false);
}

// ---------------------------------------------------------------- utilities
__global__ void zero_cnt_kernel(int* cnt) {
  if (threadIdx.x < EE) cnt[threadIdx.x] = 0;
}

__global__ void cvt_bf16_kernel(const float* __restrict__ in,
                                u16* __restrict__ out, int n) {
  int i = blockIdx.x * blockDim.x + threadIdx.x;
  if (i < n) out[i] = f2bf(in[i]);
}

// in: fp32 [R][C] (per expert) -> out: bf16 [C][R] (per expert)
__global__ void transpose_cvt_kernel(const float* __restrict__ in,
                                     u16* __restrict__ out, int R, int C) {
  __shared__ float tile[32][33];
  const size_t esz = (size_t)R * C;
  in  += esz * blockIdx.z;
  out += esz * blockIdx.z;
  const int c0 = blockIdx.x * 32, r0 = blockIdx.y * 32;
  const int tx = threadIdx.x, ty = threadIdx.y;      // block (32,8)
#pragma unroll
  for (int i = 0; i < 4; ++i)
    tile[ty + 8 * i][tx] = in[(size_t)(r0 + ty + 8 * i) * C + c0 + tx];
  __syncthreads();
#pragma unroll
  for (int i = 0; i < 4; ++i)
    out[(size_t)(c0 + ty + 8 * i) * R + r0 + tx] = f2bf(tile[tx][ty + 8 * i]);
}

// ------------------------------------------------------------------- router
// One wave per token: 8 expert scores -> top-2 -> renormalized weights.
__global__ void router_kernel(const float* __restrict__ x,
                              const float* __restrict__ wg,
                              int* __restrict__ cnt,
                              int* __restrict__ list,
                              int* __restrict__ slotk,
                              float* __restrict__ wts) {
  const int gid  = blockIdx.x * blockDim.x + threadIdx.x;
  const int t    = gid >> 5;
  const int lane = gid & 31;
  if (t >= TT) return;
  const float* xr = x + (size_t)t * HH;
  float d[EE];
#pragma unroll
  for (int e = 0; e < EE; ++e) d[e] = 0.f;
  for (int i = lane; i < HH; i += 32) {
    float xv = xr[i];
#pragma unroll
    for (int e = 0; e < EE; ++e) d[e] += xv * wg[e * HH + i];
  }
#pragma unroll
  for (int e = 0; e < EE; ++e)
    for (int off = 16; off > 0; off >>= 1)
      d[e] += __shfl_xor(d[e], off, 32);
  if (lane == 0) {
    int e0 = 0;
    for (int e = 1; e < EE; ++e) if (d[e] > d[e0]) e0 = e;          // lowest-index ties
    int e1 = (e0 == 0) ? 1 : 0;
    for (int e = 0; e < EE; ++e) if (e != e0 && d[e] > d[e1]) e1 = e;
    // topw renormalized: w0 = exp(s0)/(exp(s0)+exp(s1))
    float w0 = 1.f / (1.f + __expf(d[e1] - d[e0]));
    float w1 = 1.f - w0;
    int p0 = atomicAdd(&cnt[e0], 1);
    list[e0 * TT + p0] = t; slotk[e0 * TT + p0] = 0; wts[e0 * TT + p0] = w0;
    int p1 = atomicAdd(&cnt[e1], 1);
    list[e1 * TT + p1] = t; slotk[e1 * TT + p1] = 1; wts[e1 * TT + p1] = w1;
  }
}

// ------------------------------------------------- fused grouped expert MLP
// Block = 16 tokens of one expert, 8 waves. Phase A: act = silu(x*W1+b1) in
// LDS (bf16). Phase B: y = act*W2+b2, scaled by routing weight, scattered to
// slot-indexed ybuf (deterministic, no float atomics).
__global__ __launch_bounds__(256)
void moe_expert_kernel(const u16* __restrict__ xb,
                       const u16* __restrict__ w1t,   // [E][F][H] bf16
                       const u16* __restrict__ w2t,   // [E][H][F] bf16
                       const float* __restrict__ b1,
                       const float* __restrict__ b2,
                       const int* __restrict__ cnt,
                       const int* __restrict__ list,
                       const int* __restrict__ slotk,
                       const float* __restrict__ wts,
                       float* __restrict__ ybuf) {
  __shared__ __align__(16) u16 act[16 * FF];          // 64 KB activation tile
  const int e = blockIdx.y;
  const int c = cnt[e];
  const int mbase = blockIdx.x * 16;
  if (mbase >= c) return;

  const int tid   = threadIdx.x;
  const int w     = tid >> 5;       // wave 0..7
  const int lane  = tid & 31;
  const int hl    = lane >> 4;      // lane half (ISA A/B layout)
  const int nlane = lane & 15;
  const int eT    = e * TT;

  // Gathered A row (matrix row = lane&15); clamp rows past cnt.
  int arowi = mbase + nlane; if (arowi >= c) arowi = c - 1;
  const u16* arow = xb + (size_t)list[eT + arowi] * HH;
  const u16* w1e  = w1t + (size_t)e * FF * HH;
  const u16* w2e  = w2t + (size_t)e * HH * FF;

  // ---- Phase A: act[16][FF] = silu(x16xH * W1 + b1), bf16 in LDS ----
  for (int q = 0; q < 4; ++q) {                       // 4 groups of 4 n-tiles/wave
    v8f acc[4];
#pragma unroll
    for (int j = 0; j < 4; ++j)
#pragma unroll
      for (int r = 0; r < 8; ++r) acc[j][r] = 0.f;
    for (int ks = 0; ks < HH / 32; ++ks) {
      const int kb = ks * 32;
      BF16Frag A;
      A.q[0] = *(const uint4*)(arow + kb + 16 * hl);
      A.q[1] = *(const uint4*)(arow + kb + 16 * hl + 8);
#pragma unroll
      for (int j = 0; j < 4; ++j) {
        const int nt = w + 8 * (4 * q + j);           // n-tile index (0..127)
        const u16* brow = w1e + (size_t)(nt * 16 + nlane) * HH;
        BF16Frag B;
        B.q[0] = *(const uint4*)(brow + kb + 16 * hl);
        B.q[1] = *(const uint4*)(brow + kb + 16 * hl + 8);
        acc[j] = wmma_bf16(A.v, B.v, acc[j]);
      }
    }
#pragma unroll
    for (int j = 0; j < 4; ++j) {
      const int nt   = w + 8 * (4 * q + j);
      const int fcol = nt * 16 + nlane;
      const float bias = b1[e * FF + fcol];
#pragma unroll
      for (int r = 0; r < 8; ++r) {
        float v = acc[j][r] + bias;
        float s = v / (1.f + __expf(-v));             // silu
        act[(r + 8 * hl) * FF + fcol] = f2bf(s);
      }
    }
  }
  __syncthreads();

  // Per-lane row metadata for the scatter.
  int tokv[8]; int kv[8]; float wv[8]; bool valid[8];
#pragma unroll
  for (int r = 0; r < 8; ++r) {
    int i = mbase + r + 8 * hl;
    valid[r] = (i < c);
    int ii = valid[r] ? i : (c - 1);
    tokv[r] = list[eT + ii];
    kv[r]   = slotk[eT + ii];
    wv[r]   = wts[eT + ii];
  }

  const u16* arow2 = act + nlane * FF;                // LDS A row

  // ---- Phase B: y[16][HH] = act * W2 + b2 ----
  {
    v8f acc[4];
#pragma unroll
    for (int j = 0; j < 4; ++j)
#pragma unroll
      for (int r = 0; r < 8; ++r) acc[j][r] = 0.f;
    for (int ks = 0; ks < FF / 32; ++ks) {
      const int kb = ks * 32;
      BF16Frag A;
      A.q[0] = *(const uint4*)(arow2 + kb + 16 * hl);
      A.q[1] = *(const uint4*)(arow2 + kb + 16 * hl + 8);
#pragma unroll
      for (int j = 0; j < 4; ++j) {
        const int nt = w + 8 * j;                     // n-tile 0..31
        const u16* brow = w2e + (size_t)(nt * 16 + nlane) * FF;
        BF16Frag B;
        B.q[0] = *(const uint4*)(brow + kb + 16 * hl);
        B.q[1] = *(const uint4*)(brow + kb + 16 * hl + 8);
        acc[j] = wmma_bf16(A.v, B.v, acc[j]);
      }
    }
#pragma unroll
    for (int j = 0; j < 4; ++j) {
      const int nt   = w + 8 * j;
      const int hcol = nt * 16 + nlane;
      const float bias = b2[e * HH + hcol];
#pragma unroll
      for (int r = 0; r < 8; ++r) {
        if (valid[r]) {
          ybuf[((size_t)kv[r] * TT + tokv[r]) * HH + hcol] =
              wv[r] * (acc[j][r] + bias);
        }
      }
    }
  }
}

__global__ void combine_kernel(const float* __restrict__ y,
                               float* __restrict__ out, int n) {
  int i = blockIdx.x * blockDim.x + threadIdx.x;
  if (i < n) out[i] = y[i] + y[n + i];
}

// ------------------------------------------------------------------ launch
extern "C" void kernel_launch(void* const* d_in, const int* in_sizes, int n_in,
                              void* d_out, int out_size, void* d_ws, size_t ws_size,
                              hipStream_t stream) {
  (void)in_sizes; (void)n_in; (void)out_size; (void)ws_size;
  const float* x  = (const float*)d_in[0];
  const float* Wg = (const float*)d_in[1];
  const float* W1 = (const float*)d_in[2];
  const float* b1 = (const float*)d_in[3];
  const float* W2 = (const float*)d_in[4];
  const float* b2 = (const float*)d_in[5];
  float* out = (float*)d_out;

  // Workspace layout (~44 MB total).
  char* ws = (char*)d_ws;
  size_t o = 0;
  int*   cnt   = (int*)(ws + o);   o += 256;
  int*   list  = (int*)(ws + o);   o += (size_t)EE * TT * 4;
  int*   slotk = (int*)(ws + o);   o += (size_t)EE * TT * 4;
  float* wts   = (float*)(ws + o); o += (size_t)EE * TT * 4;
  u16*   xb    = (u16*)(ws + o);   o += (size_t)TT * HH * 2;
  u16*   w1t   = (u16*)(ws + o);   o += (size_t)EE * FF * HH * 2;
  u16*   w2t   = (u16*)(ws + o);   o += (size_t)EE * HH * FF * 2;
  float* ybuf  = (float*)(ws + o); o += (size_t)2 * TT * HH * 4;

  zero_cnt_kernel<<<1, 32, 0, stream>>>(cnt);
  cvt_bf16_kernel<<<(TT * HH) / 256, 256, 0, stream>>>(x, xb, TT * HH);
  transpose_cvt_kernel<<<dim3(FF / 32, HH / 32, EE), dim3(32, 8), 0, stream>>>(
      W1, w1t, HH, FF);
  transpose_cvt_kernel<<<dim3(HH / 32, FF / 32, EE), dim3(32, 8), 0, stream>>>(
      W2, w2t, FF, HH);
  router_kernel<<<TT / 8, 256, 0, stream>>>(x, Wg, cnt, list, slotk, wts);
  moe_expert_kernel<<<dim3(TT / 16, EE), 256, 0, stream>>>(
      xb, w1t, w2t, b1, b2, cnt, list, slotk, wts, ybuf);
  combine_kernel<<<(TT * HH) / 256, 256, 0, stream>>>(ybuf, out, TT * HH);
}
